// view_GCN_77060303225310
// MI455X (gfx1250) — matrix operand order repaired and checked by reading, not compiled, not run on hardware
//
#include <hip/hip_runtime.h>
#include <math.h>

// ---------------------------------------------------------------------------
// view-GCN forward for MI455X (gfx1250, wave32).
// All GEMMs via v_wmma_f32_16x16x32_bf16 (f32 operands converted to bf16 in
// registers, f32 accumulate).  Everything else is small scalar kernels.
// ---------------------------------------------------------------------------

typedef __bf16 bf16;
typedef __attribute__((ext_vector_type(16))) __bf16 v16bf;
typedef __attribute__((ext_vector_type(8)))  float  v8f;

__device__ __forceinline__ float lrelu(float x) { return x > 0.f ? x : 0.2f * x; }

// ---------------------------------------------------------------------------
// GEMM:  C[M,N] = act( (A[M,K] @ W[K,N] + bias) [bn] )
// flags bit0: leaky-relu epilogue, bit1: batchnorm epilogue (eval mode).
// One wave32 owns a 64(row) x 16(col) block = 4 WMMA tiles sharing one B
// fragment (B needs 16 N-strided loads; A is one contiguous run of four float4s per
// lane per tile).  K must be a multiple of 32 (true at every call site).
// Out-of-range rows/cols: loads are CLAMPED (wrong-but-safe data only feeds
// outputs that are never stored), stores are guarded -> no divergent
// branches inside the K loop.
// Fragment layouts (A, B: this kernel's contiguous k order; C per CDNA5 ISA 7.12.2):
//   A 16x32 bf16 : lane(m=lane&15, h=lane>>4), vgpr i -> K = h*16 + i*2
//   B 32x16 bf16 : lane(n=lane&15, h=lane>>4), vgpr j -> K = h*16 + j*2
//   C 16x16 f32  : vgpr r -> row = r + 8*h, col = lane&15
// ---------------------------------------------------------------------------
__global__ void gemm_wmma_bf16(const float* __restrict__ A, const float* __restrict__ W,
                               const float* __restrict__ bias,
                               const float* __restrict__ gamma, const float* __restrict__ beta,
                               const float* __restrict__ mean,  const float* __restrict__ var,
                               float* __restrict__ C, int M, int N, int K, int flags)
{
    const int lane    = threadIdx.x & 31;
    const int wid     = threadIdx.y;
    const int nTiles  = (N + 15) >> 4;
    const int mBlocks = (M + 63) >> 6;
    const int blk     = blockIdx.x * blockDim.y + wid;
    if (blk >= mBlocks * nTiles) return;
    const int mB   = blk / nTiles;
    const int nT   = blk - mB * nTiles;
    const int row0 = mB << 6;
    const int col0 = nT << 4;

    const int  mloc   = lane & 15;
    const int  hf     = lane >> 4;
    const int  colRaw = col0 + mloc;
    const bool colOK  = colRaw < N;
    const int  col    = colOK ? colRaw : (N - 1);       // clamped load column

    // clamped A row pointers for the 4 sub-tiles, offset by this lane's K run
    const float* Ar[4];
#pragma unroll
    for (int r = 0; r < 4; ++r) {
        int rowA = row0 + r * 16 + mloc;
        if (rowA >= M) rowA = M - 1;                    // clamp (store guarded)
        Ar[r] = A + (size_t)rowA * K + (hf << 4);
    }
    const float* Wcol = W + (size_t)(hf << 4) * N + col;

    v8f acc[4];
#pragma unroll
    for (int r = 0; r < 4; ++r) acc[r] = (v8f){0.f, 0.f, 0.f, 0.f, 0.f, 0.f, 0.f, 0.f};

    for (int k0 = 0; k0 < K; k0 += 32) {
        // ---- B fragment: 16 N-strided loads (coalesced across lanes) ----
        float wv[16];
        const float* Wp = Wcol + (size_t)k0 * N;
#pragma unroll
        for (int j = 0; j < 16; ++j) wv[j] = Wp[(size_t)j * N];
        v16bf b;
#pragma unroll
        for (int j = 0; j < 16; ++j) b[j] = (bf16)wv[j];

        // ---- 4 A fragments (one contiguous 16-float run each) + 4 WMMAs ----
#pragma unroll
        for (int r = 0; r < 4; ++r) {
            const float* Ap = Ar[r] + k0;
            const float4 a0 = *(const float4*)(Ap);
            const float4 a1 = *(const float4*)(Ap + 4);
            const float4 a2 = *(const float4*)(Ap + 8);
            const float4 a3 = *(const float4*)(Ap + 12);
            v16bf a;
            a[0]  = (bf16)a0.x; a[1]  = (bf16)a0.y; a[2]  = (bf16)a0.z; a[3]  = (bf16)a0.w;
            a[4]  = (bf16)a1.x; a[5]  = (bf16)a1.y; a[6]  = (bf16)a1.z; a[7]  = (bf16)a1.w;
            a[8]  = (bf16)a2.x; a[9]  = (bf16)a2.y; a[10] = (bf16)a2.z; a[11] = (bf16)a2.w;
            a[12] = (bf16)a3.x; a[13] = (bf16)a3.y; a[14] = (bf16)a3.z; a[15] = (bf16)a3.w;
            acc[r] = __builtin_amdgcn_wmma_f32_16x16x32_bf16(false, a, false, b,
                                                             (short)0, acc[r], false, false);
        }
        if (k0 + 32 < K) {
            __builtin_prefetch(Ar[0] + k0 + 32, 0, 1);
            __builtin_prefetch(Wcol + (size_t)(k0 + 32) * N, 0, 1);
        }
    }

    if (!colOK) return;
    float sc = 1.f;
    float sh = bias ? bias[colRaw] : 0.f;
    if (flags & 2) {
        const float inv = rsqrtf(var[colRaw] + 1e-5f);
        sc = gamma[colRaw] * inv;
        sh = (sh - mean[colRaw]) * sc + beta[colRaw];
    }
#pragma unroll
    for (int r = 0; r < 4; ++r) {
        const int rowBase = row0 + r * 16 + (hf << 3);
#pragma unroll
        for (int q = 0; q < 8; ++q) {
            const int row = rowBase + q;
            if (row < M) {
                float v = acc[r][q] * sc + sh;
                if (flags & 1) v = lrelu(v);
                C[(size_t)row * N + colRaw] = v;
            }
        }
    }
}

// --------------------------- scalar helper kernels -------------------------

// broadcast the [20,3] vertex table to [B,20,3]
__global__ void bcast_verts_kernel(const float* __restrict__ v0, float* __restrict__ out, int total)
{
    int t = blockIdx.x * blockDim.x + threadIdx.x;
    if (t < total) out[t] = v0[t % 60];
}

// knn (k=4) with reference's -2ab + |a|^2 + |b|^2 metric, stable ties.
__global__ void knn_kernel(const float* __restrict__ src, const float* __restrict__ dst,
                           int B, int N, int Mq, int* __restrict__ idx)
{
    int t = blockIdx.x * blockDim.x + threadIdx.x;
    if (t >= B * Mq) return;
    int b = t / Mq, m = t - b * Mq;
    const float* S = src + (size_t)b * N * 3;
    const float* D = dst + (size_t)b * Mq * 3;
    float dx = D[m * 3], dy = D[m * 3 + 1], dz = D[m * 3 + 2];
    float dd = dx * dx + dy * dy + dz * dz;
    float bestd[4]; int besti[4];
#pragma unroll
    for (int j = 0; j < 4; ++j) { bestd[j] = 1e30f; besti[j] = 0; }
    for (int n = 0; n < N; ++n) {
        float sx = S[n * 3], sy = S[n * 3 + 1], sz = S[n * 3 + 2];
        float d = -2.f * (sx * dx + sy * dy + sz * dz) + (sx * sx + sy * sy + sz * sz) + dd;
        int j = 4;
        while (j > 0 && d < bestd[j - 1]) --j;
        if (j < 4) {
            for (int q = 3; q > j; --q) { bestd[q] = bestd[q - 1]; besti[q] = besti[q - 1]; }
            bestd[j] = d; besti[j] = n;
        }
    }
#pragma unroll
    for (int j = 0; j < 4; ++j) idx[(size_t)t * 4 + j] = besti[j];
}

// KNN_dist edge-MLP (10->10->10->1, leaky between) -> per-neighbor weight.
__global__ void knnw_kernel(const float* __restrict__ verts, const int* __restrict__ idx,
                            const float* __restrict__ r1W, const float* __restrict__ r1b,
                            const float* __restrict__ r2W, const float* __restrict__ r2b,
                            const float* __restrict__ r3W, const float* __restrict__ r3b,
                            int B, int N, float* __restrict__ wout)
{
    int t = blockIdx.x * blockDim.x + threadIdx.x;
    if (t >= B * N) return;
    int b = t / N;
    const float* V = verts + (size_t)b * N * 3;
    const int* I = idx + (size_t)t * 4;
    float vk[4][3];
#pragma unroll
    for (int k = 0; k < 4; ++k) {
        int nn = I[k];
        vk[k][0] = V[nn * 3]; vk[k][1] = V[nn * 3 + 1]; vk[k][2] = V[nn * 3 + 2];
    }
    float v0x = vk[0][0], v0y = vk[0][1], v0z = vk[0][2];
#pragma unroll
    for (int k = 0; k < 4; ++k) {
        float d0 = v0x - vk[k][0], d1 = v0y - vk[k][1], d2 = v0z - vk[k][2];
        float nr = sqrtf(fmaxf(d0 * d0 + d1 * d1 + d2 * d2, 1e-12f));
        float x[10] = {v0x, v0y, v0z, vk[k][0], vk[k][1], vk[k][2], d0, d1, d2, nr};
        float h1[10], h2[10];
#pragma unroll
        for (int o = 0; o < 10; ++o) {
            float s = r1b[o];
            for (int i = 0; i < 10; ++i) s += x[i] * r1W[i * 10 + o];
            h1[o] = lrelu(s);
        }
#pragma unroll
        for (int o = 0; o < 10; ++o) {
            float s = r2b[o];
            for (int i = 0; i < 10; ++i) s += h1[i] * r2W[i * 10 + o];
            h2[o] = lrelu(s);
        }
        float s = r3b[0];
        for (int i = 0; i < 10; ++i) s += h2[i] * r3W[i];
        wout[(size_t)t * 4 + k] = s;
    }
}

// F_out[b,i,c] = sum_k w[b,i,k] * F[b, idx[b,i,k], c]
__global__ void wsum_kernel(const float* __restrict__ F, const int* __restrict__ idx,
                            const float* __restrict__ w, int B, int N, float* __restrict__ out)
{
    int t = blockIdx.x * blockDim.x + threadIdx.x;
    if (t >= B * N * 512) return;
    int c = t & 511, u = t >> 9;
    int i = u % N, b = u / N;
    const int*   I = idx + ((size_t)b * N + i) * 4;
    const float* W = w   + ((size_t)b * N + i) * 4;
    float s = 0.f;
#pragma unroll
    for (int k = 0; k < 4; ++k) s += W[k] * F[((size_t)b * N + I[k]) * 512 + c];
    out[t] = s;
}

// M1[b,i,j,c] = leaky(Pi[b,i,c] + Pj[b,j,c])   (factored nonlocal r1)
__global__ void nlcomb_kernel(const float* __restrict__ Pi, const float* __restrict__ Pj,
                              int B, int n, float* __restrict__ out)
{
    int t = blockIdx.x * blockDim.x + threadIdx.x;
    if (t >= B * n * n * 512) return;
    int c = t & 511, u = t >> 9;
    int j = u % n; u /= n;
    int i = u % n; int b = u / n;
    out[t] = lrelu(Pi[((size_t)b * n + i) * 512 + c] + Pj[((size_t)b * n + j) * 512 + c]);
}

// Msum[b,i,c] = sum_j M3[b,i,j,c]
__global__ void sumj_kernel(const float* __restrict__ M3, int B, int n, float* __restrict__ out)
{
    int t = blockIdx.x * blockDim.x + threadIdx.x;
    if (t >= B * n * 512) return;
    int c = t & 511, u = t >> 9;
    int i = u % n, b = u / n;
    const float* base = M3 + (((size_t)b * n + i) * n) * 512 + c;
    float s = 0.f;
    for (int j = 0; j < n; ++j) s += base[(size_t)j * 512];
    out[t] = s;
}

// cat[b,i,:] = [A[b,i,:512], B[b,i,:512]]
__global__ void concat2_kernel(const float* __restrict__ A, const float* __restrict__ Bs,
                               int B, int n, float* __restrict__ out)
{
    int t = blockIdx.x * blockDim.x + threadIdx.x;
    if (t >= B * n * 1024) return;
    int c = t & 1023, u = t >> 10;
    int i = u % n, b = u / n;
    size_t r = (size_t)b * n + i;
    out[t] = (c < 512) ? A[r * 512 + c] : Bs[r * 512 + (c - 512)];
}

// pooled[b, off+c] = max_i X[b,i,c]
__global__ void maxpool_kernel(const float* __restrict__ X, int B, int n,
                               float* __restrict__ pooled, int off)
{
    int t = blockIdx.x * blockDim.x + threadIdx.x;
    if (t >= B * 512) return;
    int c = t & 511, b = t >> 9;
    float m = -1e30f;
    for (int i = 0; i < n; ++i) m = fmaxf(m, X[((size_t)b * n + i) * 512 + c]);
    pooled[(size_t)b * 1536 + off + c] = m;
}

// farthest point sampling, deterministic start at index 0 (scan semantics).
__global__ void fps_kernel(const float* __restrict__ verts, int B, int N, int S, int* __restrict__ out)
{
    int b = blockIdx.x * blockDim.x + threadIdx.x;
    if (b >= B) return;
    const float* V = verts + (size_t)b * N * 3;
    float dist[20];
    for (int n = 0; n < N; ++n) dist[n] = 1e10f;
    int far = 0;
    for (int t = 0; t < S; ++t) {
        out[(size_t)b * S + t] = far;
        float cx = V[far * 3], cy = V[far * 3 + 1], cz = V[far * 3 + 2];
        float best = -1e30f; int bi = 0;
        for (int n = 0; n < N; ++n) {
            float dx = V[n * 3] - cx, dy = V[n * 3 + 1] - cy, dz = V[n * 3 + 2] - cz;
            float d = dx * dx + dy * dy + dz * dz;
            if (d < dist[n]) dist[n] = d;
            if (dist[n] > best) { best = dist[n]; bi = n; }
        }
        far = bi;
    }
}

// v1[b,s,:] = verts[b, fpsidx[b,s], :]
__global__ void gather_v1_kernel(const float* __restrict__ verts, const int* __restrict__ idx,
                                 int B, int N, int S, float* __restrict__ v1)
{
    int t = blockIdx.x * blockDim.x + threadIdx.x;
    if (t >= B * S * 3) return;
    int d = t % 3, u = t / 3;
    int s = u % S, b = u / S;
    v1[t] = verts[((size_t)b * N + idx[(size_t)b * S + s]) * 3 + d];
}

// Fk[b,s,k,c]=F[b,idk,c];  vk likewise (c<3);  F1[b,k, s*512+c]=Fk (transposed view)
__global__ void vs_gather_kernel(const float* __restrict__ F, const float* __restrict__ verts,
                                 const int* __restrict__ idk, int B, int N, int S,
                                 float* __restrict__ Fk, float* __restrict__ vk, float* __restrict__ F1)
{
    int t = blockIdx.x * blockDim.x + threadIdx.x;
    if (t >= B * S * 4 * 512) return;
    int c = t & 511, u = t >> 9;
    int k = u & 3; u >>= 2;
    int s = u % S, b = u / S;
    int nn = idk[(((size_t)b * S + s)) * 4 + k];
    float v = F[((size_t)b * N + nn) * 512 + c];
    Fk[t] = v;
    F1[((size_t)b * 4 + k) * ((size_t)S * 512) + (size_t)s * 512 + c] = v;
    if (c < 3) vk[(((size_t)b * S + s) * 4 + k) * 3 + c] = verts[((size_t)b * N + nn) * 3 + c];
}

// Fscore[b,s,k,c] = score[b,k, s*40+c]
__global__ void fscore_scatter_kernel(const float* __restrict__ score, int B, int S,
                                      float* __restrict__ Fs)
{
    int t = blockIdx.x * blockDim.x + threadIdx.x;
    if (t >= B * S * 160) return;
    int c = t % 40, u = t / 40;
    int k = u & 3; u >>= 2;
    int s = u % S, b = u / S;
    Fs[t] = score[((size_t)b * 4 + k) * ((size_t)S * 40) + (size_t)s * 40 + c];
}

// softmax over s: per (b,k,c) compute max and denom
__global__ void softmax_stats_kernel(const float* __restrict__ Fs, int B, int S,
                                     float* __restrict__ mx, float* __restrict__ den)
{
    int t = blockIdx.x * blockDim.x + threadIdx.x;
    if (t >= B * 160) return;
    int c = t % 40, u = t / 40;
    int k = u & 3, b = u >> 2;
    float m = -1e30f;
    for (int s = 0; s < S; ++s) m = fmaxf(m, Fs[(((size_t)b * S + s) * 4 + k) * 40 + c]);
    float d = 0.f;
    for (int s = 0; s < S; ++s) d += expf(Fs[(((size_t)b * S + s) * 4 + k) * 40 + c] - m);
    mx[t] = m; den[t] = d;
}

// prob[b,s,k] = max_c softmax;  sel[b,s] = argmax_k prob (first max)
__global__ void argsel_kernel(const float* __restrict__ Fs, const float* __restrict__ mx,
                              const float* __restrict__ den, int B, int S, int* __restrict__ sel)
{
    int t = blockIdx.x * blockDim.x + threadIdx.x;
    if (t >= B * S) return;
    int b = t / S, s = t - b * S;
    float bestp = -1e30f; int bk = 0;
    for (int k = 0; k < 4; ++k) {
        float pm = -1e30f;
        for (int c = 0; c < 40; ++c) {
            size_t si = ((size_t)b * 4 + k) * 40 + c;
            float e = expf(Fs[(((size_t)b * S + s) * 4 + k) * 40 + c] - mx[si]) / den[si];
            pm = fmaxf(pm, e);
        }
        if (pm > bestp) { bestp = pm; bk = k; }
    }
    sel[t] = bk;
}

// F_new[b,s,:]=Fk[b,s,sel,:];  v_new[b,s,:]=vk[b,s,sel,:]
__global__ void select_kernel(const float* __restrict__ Fk, const float* __restrict__ vk,
                              const int* __restrict__ sel, int B, int S,
                              float* __restrict__ Fn, float* __restrict__ vn)
{
    int t = blockIdx.x * blockDim.x + threadIdx.x;
    if (t >= B * S * 512) return;
    int c = t & 511, u = t >> 9;
    int s = u % S, b = u / S;
    int k = sel[(size_t)b * S + s];
    size_t r = (size_t)b * S + s;
    Fn[t] = Fk[(r * 4 + k) * 512 + c];
    if (c < 3) vn[r * 3 + c] = vk[(r * 4 + k) * 3 + c];
}

// --------------------------------- host ------------------------------------

static inline int cdiv(int a, int b) { return (a + b - 1) / b; }

static void launch_gemm(const float* A, const float* W, const float* bias,
                        const float* gamma, const float* beta,
                        const float* mean, const float* var,
                        float* C, int M, int N, int K, int flags, hipStream_t s)
{
    const int waves = cdiv(M, 64) * cdiv(N, 16);
    gemm_wmma_bf16<<<dim3(cdiv(waves, 8)), dim3(32, 8), 0, s>>>(
        A, W, bias, gamma, beta, mean, var, C, M, N, K, flags);
}

#define EW(n) dim3(((n) + 255) / 256), dim3(256), 0, stream

extern "C" void kernel_launch(void* const* d_in, const int* in_sizes, int n_in,
                              void* d_out, int out_size, void* d_ws, size_t ws_size,
                              hipStream_t stream)
{
    const float* const* in = (const float* const*)d_in;
    // jax.tree_leaves order (sorted dict keys at every level):
    //   params: cls1(W,b) cls2 cls3 | lg1..lg3 (bn.beta,bn.gamma,bn.mean,bn.var,
    //   conv.W,conv.b,r1.W,r1.b,r2.W,r2.b,r3.W,r3.b) | nl1,nl2 (bn*4,fusion.W,
    //   fusion.b,r1*,r2*,r3*) | vs1(c1.W,c1.b,c2.W,c2.b) vs2 ; vertices=74 ; y=75
    const int CLS1 = 0, CLS2 = 2, CLS3 = 4;
    const int LG1 = 6, LG2 = 18, LG3 = 30;
    const int NL1 = 42, NL2 = 54;
    const int VS1 = 66, VS2 = 70;
    const int IDX_VERT = 74, IDX_Y = 75;

    const int B = in_sizes[IDX_Y] / (20 * 512);   // 128
    const float* yin = in[IDX_Y];
    float* out = (float*)d_out;
    float* Fs1out = out + (size_t)B * 40;                       // [B,10,4,40]
    float* Fs2out = Fs1out + (size_t)B * 10 * 4 * 40;           // [B,5,4,40]

    // ---- bump allocator over d_ws ----
    char* bp = (char*)d_ws;
    auto allocF = [&](size_t n) -> float* {
        float* r = (float*)bp; bp += ((n * sizeof(float) + 255) & ~(size_t)255); return r;
    };
    auto allocI = [&](size_t n) -> int* {
        int* r = (int*)bp; bp += ((n * sizeof(int) + 255) & ~(size_t)255); return r;
    };

    float* verts20 = allocF((size_t)B * 60);
    float* y1   = allocF((size_t)B * 20 * 512);
    float* y2   = allocF((size_t)B * 20 * 512);
    float* tmpF = allocF((size_t)B * 20 * 512);
    float* Pi   = allocF((size_t)B * 20 * 512);
    float* Pj   = allocF((size_t)B * 20 * 512);
    float* BIG0 = allocF((size_t)B * 20 * 20 * 512);
    float* BIG1 = allocF((size_t)B * 20 * 20 * 512);
    float* Msum = allocF((size_t)B * 20 * 512);
    float* catB = allocF((size_t)B * 20 * 1024);
    int*   knnI = allocI((size_t)B * 20 * 4);
    float* knnW = allocF((size_t)B * 20 * 4);
    int*   fpsI = allocI((size_t)B * 10);
    float* v1b  = allocF((size_t)B * 10 * 3);
    int*   idk  = allocI((size_t)B * 10 * 4);
    float* Fk   = allocF((size_t)B * 10 * 4 * 512);
    float* vk   = allocF((size_t)B * 10 * 4 * 3);
    float* F1   = allocF((size_t)B * 4 * 5120);
    float* H1   = allocF((size_t)B * 4 * 2560);
    float* scoreB = allocF((size_t)B * 4 * 400);
    float* smax = allocF((size_t)B * 160);
    float* sden = allocF((size_t)B * 160);
    int*   selB = allocI((size_t)B * 10);
    float* zF   = allocF((size_t)B * 10 * 512);
    float* v2   = allocF((size_t)B * 10 * 3);
    float* z1   = allocF((size_t)B * 10 * 512);
    float* z2   = allocF((size_t)B * 10 * 512);
    float* wF   = allocF((size_t)B * 5 * 512);
    float* v3   = allocF((size_t)B * 5 * 3);
    float* w1   = allocF((size_t)B * 5 * 512);
    float* pooled = allocF((size_t)B * 1536);
    float* hc1  = allocF((size_t)B * 512);
    float* hc2  = allocF((size_t)B * 512);
    (void)ws_size; (void)n_in; (void)out_size;

    // ---- stage helpers ----
    auto local_gcn = [&](const float* F, const float* verts, int n, int LG, float* outF) {
        knn_kernel<<<EW(B * n)>>>(verts, verts, B, n, n, knnI);
        knnw_kernel<<<EW(B * n)>>>(verts, knnI, in[LG + 6], in[LG + 7], in[LG + 8],
                                   in[LG + 9], in[LG + 10], in[LG + 11], B, n, knnW);
        wsum_kernel<<<EW(B * n * 512)>>>(F, knnI, knnW, B, n, tmpF);
        // conv + bn + leaky
        launch_gemm(tmpF, in[LG + 4], in[LG + 5], in[LG + 1], in[LG + 0], in[LG + 2],
                    in[LG + 3], outF, B * n, 512, 512, 3, stream);
    };

    auto nonlocal_mp = [&](const float* F, int n, int NL, float* outF) {
        const float* r1W = in[NL + 6];
        // factored r1: Pi = F@Wi (no bias), Pj = F@Wj + b
        launch_gemm(F, r1W,             nullptr,    nullptr, nullptr, nullptr, nullptr,
                    Pi, B * n, 512, 512, 0, stream);
        launch_gemm(F, r1W + 512 * 512, in[NL + 7], nullptr, nullptr, nullptr, nullptr,
                    Pj, B * n, 512, 512, 0, stream);
        nlcomb_kernel<<<EW(B * n * n * 512)>>>(Pi, Pj, B, n, BIG0);
        launch_gemm(BIG0, in[NL + 8],  in[NL + 9],  nullptr, nullptr, nullptr, nullptr,
                    BIG1, B * n * n, 512, 512, 1, stream);
        launch_gemm(BIG1, in[NL + 10], in[NL + 11], nullptr, nullptr, nullptr, nullptr,
                    BIG0, B * n * n, 512, 512, 1, stream);
        sumj_kernel<<<EW(B * n * 512)>>>(BIG0, B, n, Msum);
        concat2_kernel<<<EW(B * n * 1024)>>>(F, Msum, B, n, catB);
        // fusion + bn + leaky
        launch_gemm(catB, in[NL + 4], in[NL + 5], in[NL + 1], in[NL + 0], in[NL + 2],
                    in[NL + 3], outF, B * n, 512, 1024, 3, stream);
    };

    auto view_sel = [&](const float* F, const float* verts, int n, int S, int VS,
                        float* FsOut, float* Fnew, float* vnew) {
        fps_kernel<<<EW(B)>>>(verts, B, n, S, fpsI);
        gather_v1_kernel<<<EW(B * S * 3)>>>(verts, fpsI, B, n, S, v1b);
        knn_kernel<<<EW(B * S)>>>(verts, v1b, B, n, S, idk);
        vs_gather_kernel<<<EW(B * S * 4 * 512)>>>(F, verts, idk, B, n, S, Fk, vk, F1);
        launch_gemm(F1, in[VS + 0], in[VS + 1], nullptr, nullptr, nullptr, nullptr,
                    H1, B * 4, 256 * S, 512 * S, 1, stream);
        launch_gemm(H1, in[VS + 2], in[VS + 3], nullptr, nullptr, nullptr, nullptr,
                    scoreB, B * 4, 40 * S, 256 * S, 0, stream);
        fscore_scatter_kernel<<<EW(B * S * 160)>>>(scoreB, B, S, FsOut);
        softmax_stats_kernel<<<EW(B * 160)>>>(FsOut, B, S, smax, sden);
        argsel_kernel<<<EW(B * S)>>>(FsOut, smax, sden, B, S, selB);
        select_kernel<<<EW(B * S * 512)>>>(Fk, vk, selB, B, S, Fnew, vnew);
    };

    // ---- pipeline ----
    bcast_verts_kernel<<<EW(B * 60)>>>(in[IDX_VERT], verts20, B * 60);

    local_gcn(yin, verts20, 20, LG1, y1);
    maxpool_kernel<<<EW(B * 512)>>>(y1, B, 20, pooled, 0);
    nonlocal_mp(y1, 20, NL1, y2);
    view_sel(y2, verts20, 20, 10, VS1, Fs1out, zF, v2);

    local_gcn(zF, v2, 10, LG2, z1);
    maxpool_kernel<<<EW(B * 512)>>>(z1, B, 10, pooled, 512);
    nonlocal_mp(z1, 10, NL2, z2);
    view_sel(z2, v2, 10, 5, VS2, Fs2out, wF, v3);

    local_gcn(wF, v3, 5, LG3, w1);
    maxpool_kernel<<<EW(B * 512)>>>(w1, B, 5, pooled, 1024);

    // classifier head
    launch_gemm(pooled, in[CLS1], in[CLS1 + 1], nullptr, nullptr, nullptr, nullptr,
                hc1, B, 512, 1536, 1, stream);
    launch_gemm(hc1, in[CLS2], in[CLS2 + 1], nullptr, nullptr, nullptr, nullptr,
                hc2, B, 512, 512, 1, stream);
    launch_gemm(hc2, in[CLS3], in[CLS3 + 1], nullptr, nullptr, nullptr, nullptr,
                out, B, 40, 512, 0, stream);
}
